// TransformerBlock_38053410242735
// MI455X (gfx1250) — matrix operand order repaired and checked by reading, not compiled, not run on hardware
//
#include <hip/hip_runtime.h>
#include <hip/hip_bf16.h>
#include <math.h>

typedef __attribute__((ext_vector_type(16))) __bf16 v16bf;
typedef __attribute__((ext_vector_type(8)))  float  v8f;

#define HCC 96
#define TILE_M 64
#define XS_STRIDE_F 132   // floats per LDS row (128 + 4 pad)

// ---- order-preserving float <-> uint for atomicMax-based segment max ----
__device__ __forceinline__ unsigned ordered_enc(float f) {
  unsigned u = __float_as_uint(f);
  return (u & 0x80000000u) ? ~u : (u | 0x80000000u);
}
__device__ __forceinline__ float ordered_dec(unsigned u) {
  unsigned v = (u & 0x80000000u) ? (u & 0x7fffffffu) : ~u;
  return __uint_as_float(v);
}
#define ENC_NEG_INF 0x007fffffu  // ordered_enc(-INF)

// ============================================================
// K1: pack Wq|Wk|Wv|Wskip (128x96 f32, row major) into bf16
// B-fragment layout: frag=(mat,nt,kt); lane l -> col n=nt*16+(l&15),
// 16 consecutive k starting at kt*32+(l>>4)*16. 32B contiguous per lane.
// ============================================================
__global__ __launch_bounds__(256) void pack_weights_kernel(
    const float* __restrict__ Wq, const float* __restrict__ Wk,
    const float* __restrict__ Wv, const float* __restrict__ Wskip,
    __bf16* __restrict__ packed) {
  int idx = blockIdx.x * 256 + threadIdx.x;
  if (idx >= 4 * 128 * 96) return;
  int j   = idx & 15;
  int l   = (idx >> 4) & 31;
  int kt  = (idx >> 9) & 3;
  int nt  = (idx >> 11) % 6;
  int mat = idx / 12288;
  int k = kt * 32 + (l >> 4) * 16 + j;
  int n = nt * 16 + (l & 15);
  const float* W = (mat == 0) ? Wq : (mat == 1) ? Wk : (mat == 2) ? Wv : Wskip;
  packed[idx] = (__bf16)W[k * 96 + n];
}

// ============================================================
// K-init: zero accumulators, -inf segment max. Runs every launch.
// ============================================================
__global__ __launch_bounds__(256) void init_kernel(
    float* __restrict__ outr, unsigned* __restrict__ amax,
    float* __restrict__ denom, float* __restrict__ sums, int N) {
  int idx = blockIdx.x * 256 + threadIdx.x;
  if (idx < N * HCC) outr[idx] = 0.f;
  if (idx < N * 3) { amax[idx] = ENC_NEG_INF; denom[idx] = 0.f; }
  if (idx < 64) sums[idx] = 0.f;
}

// ============================================================
// K2: q/k/v/x_r projections. 64 nodes per WG (4 waves, wave w owns
// m-tile w). x tile staged in LDS as f32 via async-LDS DMA on the
// full-tile fast path; bf16 conversion happens while building A
// fragments. 96 wmma per wave, f32 accumulate.
// ============================================================
__global__ __launch_bounds__(128) void proj_wmma_kernel(
    const float* __restrict__ x, const __bf16* __restrict__ packed,
    const float* __restrict__ bq, const float* __restrict__ bk,
    const float* __restrict__ bv, const float* __restrict__ bskip,
    float* __restrict__ q, float* __restrict__ k,
    float* __restrict__ v, float* __restrict__ xr, int N) {
  __shared__ float xs[TILE_M * XS_STRIDE_F];  // 33,792 B; only LDS object
  const int tid  = threadIdx.x;
  const int tile = blockIdx.x;
  const bool full = (tile + 1) * TILE_M <= N;

  if (full) {
    // Async DMA: 64 rows x 512B = 32KB, 16B per issue, ASYNCcnt-tracked.
#pragma unroll
    for (int it = 0; it < 16; ++it) {
      int idx4 = it * 128 + tid;  // 2048 float4 slots = 64 rows x 32
      int row  = idx4 >> 5;
      int c4   = idx4 & 31;
      unsigned lds_off =
          (unsigned)((row * XS_STRIDE_F + c4 * 4) * (int)sizeof(float));
      unsigned long long gp = (unsigned long long)(uintptr_t)(
          x + (size_t)(tile * TILE_M + row) * 128 + c4 * 4);
      asm volatile("global_load_async_to_lds_b128 %0, %1, off"
                   :: "v"(lds_off), "v"(gp) : "memory");
    }
    // prefetch next tile's x rows while DMA completes
    {
      int node2 = (tile + 1) * TILE_M + (tid >> 1);
      if (node2 < N)
        __builtin_prefetch(x + (size_t)node2 * 128 + (tid & 1) * 64, 0, 0);
    }
    asm volatile("s_wait_asynccnt 0" ::: "memory");
  } else {
    // tail tile: guarded loads, zero padding
#pragma unroll
    for (int it = 0; it < 16; ++it) {
      int idx4 = it * 128 + tid;
      int row  = idx4 >> 5;
      int c4   = idx4 & 31;
      int node = tile * TILE_M + row;
      float4 val = make_float4(0.f, 0.f, 0.f, 0.f);
      if (node < N)
        val = *reinterpret_cast<const float4*>(x + (size_t)node * 128 + c4 * 4);
      float* p = xs + row * XS_STRIDE_F + c4 * 4;
      p[0] = val.x; p[1] = val.y; p[2] = val.z; p[3] = val.w;
    }
  }
  __syncthreads();

  const int wave  = tid >> 5;
  const int lane  = tid & 31;
  const int ml    = lane & 15;
  const int g     = lane >> 4;
  const int mrow0 = wave * 16;

  // A fragments (contiguous k order, matching packed B):
  // lanes 0-15: K=0..15 (elems 0-15); lanes 16-31: K=16..31
  v16bf a[4];
  const float* xrow = xs + (mrow0 + ml) * XS_STRIDE_F;
#pragma unroll
  for (int kt = 0; kt < 4; ++kt) {
#pragma unroll
    for (int j = 0; j < 16; ++j)
      a[kt][j] = (__bf16)xrow[kt * 32 + g * 16 + j];
  }

  float* dsts[4]       = {q, k, v, xr};
  const float* bias[4] = {bq, bk, bv, bskip};

#pragma unroll
  for (int mat = 0; mat < 4; ++mat) {
#pragma unroll
    for (int nt = 0; nt < 6; ++nt) {
      v8f acc = {};
#pragma unroll
      for (int kt = 0; kt < 4; ++kt) {
        const v16bf b = *reinterpret_cast<const v16bf*>(
            packed + ((((mat * 6 + nt) * 4 + kt) * 32 + lane) << 4));
        acc = __builtin_amdgcn_wmma_f32_16x16x32_bf16(
            false, a[kt], false, b, (short)0, acc, false, false);
      }
      const float bn = bias[mat][nt * 16 + ml];
      float* dst = dsts[mat];
      // D layout: VGPR r -> (M = r + 8*g, N = ml)
      const size_t base =
          (size_t)(tile * TILE_M + mrow0 + g * 8) * HCC + nt * 16 + ml;
      if (full) {
#pragma unroll
        for (int r = 0; r < 8; ++r)
          dst[base + (size_t)r * HCC] = acc[r] + bn;
      } else {
#pragma unroll
        for (int r = 0; r < 8; ++r) {
          int node = tile * TILE_M + mrow0 + g * 8 + r;
          if (node < N) dst[(size_t)node * HCC + nt * 16 + ml] = acc[r] + bn;
        }
      }
    }
  }
}

// ============================================================
// K3a: per-edge attention logits + segment max (ordered-uint atomicMax)
// ============================================================
__global__ __launch_bounds__(256) void edge_alpha_kernel(
    const float* __restrict__ q, const float* __restrict__ k,
    const float* __restrict__ edge_attr, const float* __restrict__ We,
    const int* __restrict__ eidx, float* __restrict__ alpha,
    unsigned* __restrict__ amax, int E) {
  __shared__ float sWe[96];
  if (threadIdx.x < 96) sWe[threadIdx.x] = We[threadIdx.x];
  __syncthreads();
  int e = blockIdx.x * 256 + threadIdx.x;
  if (e >= E) return;
  int src = eidx[e];
  int dst = eidx[E + e];
  float ea = edge_attr[e];
  const float4* qp = reinterpret_cast<const float4*>(q + (size_t)dst * HCC);
  const float4* kp = reinterpret_cast<const float4*>(k + (size_t)src * HCC);
  const float4* wp = reinterpret_cast<const float4*>(sWe);
  float acc[3] = {0.f, 0.f, 0.f};
#pragma unroll
  for (int j4 = 0; j4 < 24; ++j4) {
    float4 qv = qp[j4], kv = kp[j4], wv = wp[j4];
    acc[j4 >> 3] += qv.x * (kv.x + ea * wv.x) + qv.y * (kv.y + ea * wv.y)
                  + qv.z * (kv.z + ea * wv.z) + qv.w * (kv.w + ea * wv.w);
  }
  const float scale = 0.17677669529663689f;  // 1/sqrt(32)
#pragma unroll
  for (int h = 0; h < 3; ++h) {
    float al = acc[h] * scale;
    alpha[(size_t)e * 3 + h] = al;
    atomicMax(&amax[(size_t)dst * 3 + h], ordered_enc(al));
  }
}

// ============================================================
// K3b: exp-weights, denom atomicAdd, scatter-add of weighted v+e.
// Division by denom deferred to node kernel (legal: denom is per-dst).
// ============================================================
__global__ __launch_bounds__(256) void edge_accum_kernel(
    const float* __restrict__ v, const float* __restrict__ edge_attr,
    const float* __restrict__ We, const int* __restrict__ eidx,
    const float* __restrict__ alpha, const unsigned* __restrict__ amax,
    float* __restrict__ denom, float* __restrict__ outr, int E) {
  __shared__ float sWe[96];
  if (threadIdx.x < 96) sWe[threadIdx.x] = We[threadIdx.x];
  __syncthreads();
  int e = blockIdx.x * 256 + threadIdx.x;
  if (e >= E) return;
  int src = eidx[e];
  int dst = eidx[E + e];
  float ea = edge_attr[e];
  float w[3];
#pragma unroll
  for (int h = 0; h < 3; ++h) {
    float m  = ordered_dec(amax[(size_t)dst * 3 + h]);
    float wv = __expf(alpha[(size_t)e * 3 + h] - m);
    w[h] = wv;
    atomicAdd(&denom[(size_t)dst * 3 + h], wv);
  }
  const float4* vp = reinterpret_cast<const float4*>(v + (size_t)src * HCC);
  const float4* wp = reinterpret_cast<const float4*>(sWe);
  float* op = outr + (size_t)dst * HCC;
#pragma unroll
  for (int j4 = 0; j4 < 24; ++j4) {
    float4 vv = vp[j4], wv = wp[j4];
    float wh = w[j4 >> 3];
    atomicAdd(op + j4 * 4 + 0, (vv.x + ea * wv.x) * wh);
    atomicAdd(op + j4 * 4 + 1, (vv.y + ea * wv.y) * wh);
    atomicAdd(op + j4 * 4 + 2, (vv.z + ea * wv.z) * wh);
    atomicAdd(op + j4 * 4 + 3, (vv.w + ea * wv.w) * wh);
  }
}

// ============================================================
// K4: per-node epilogue: softmax denom, beta gate, 96->32 linear,
// ReLU, store h, LDS-reduced BN sums -> global atomics.
// ============================================================
__global__ __launch_bounds__(256) void node_finalize_kernel(
    const float* __restrict__ outr, const float* __restrict__ denom,
    const float* __restrict__ xr, const float* __restrict__ Wbeta,
    const float* __restrict__ Wt, const float* __restrict__ bt,
    float* __restrict__ hbuf, float* __restrict__ sums, int N) {
  __shared__ float sWt[96 * 32];
  __shared__ float sWb[288];
  __shared__ float sbt[32];
  __shared__ float sSum[32];
  __shared__ float sSq[32];
  for (int i = threadIdx.x; i < 96 * 32; i += 256) sWt[i] = Wt[i];
  for (int i = threadIdx.x; i < 288; i += 256) sWb[i] = Wbeta[i];
  if (threadIdx.x < 32) {
    sbt[threadIdx.x] = bt[threadIdx.x];
    sSum[threadIdx.x] = 0.f;
    sSq[threadIdx.x] = 0.f;
  }
  __syncthreads();

  int n = blockIdx.x * 256 + threadIdx.x;
  if (n < N) {
    float dinv[3];
#pragma unroll
    for (int h = 0; h < 3; ++h)
      dinv[h] = 1.f / (denom[(size_t)n * 3 + h] + 1e-16f);
    const float* orow = outr + (size_t)n * HCC;
    const float* rrow = xr + (size_t)n * HCC;

    float bacc = 0.f;
#pragma unroll
    for (int h = 0; h < 3; ++h) {
      float di = dinv[h];
      for (int j = h * 32; j < h * 32 + 32; ++j) {
        float o = orow[j] * di;
        float r = rrow[j];
        bacc += sWb[j] * o + sWb[96 + j] * r + sWb[192 + j] * (o - r);
      }
    }
    float beta = 1.f / (1.f + __expf(-bacc));

    float hacc[32];
#pragma unroll
    for (int t = 0; t < 32; ++t) hacc[t] = sbt[t];
#pragma unroll
    for (int h = 0; h < 3; ++h) {
      float di = dinv[h];
      for (int j = h * 32; j < h * 32 + 32; ++j) {
        float o  = orow[j] * di;
        float r  = rrow[j];
        float gv = beta * r + (1.f - beta) * o;
#pragma unroll
        for (int t = 0; t < 32; ++t)
          hacc[t] = __fmaf_rn(gv, sWt[j * 32 + t], hacc[t]);
      }
    }
#pragma unroll
    for (int t = 0; t < 32; ++t) {
      float hv = fmaxf(hacc[t], 0.f);
      hbuf[(size_t)n * 32 + t] = hv;
      atomicAdd(&sSum[t], hv);
      atomicAdd(&sSq[t], hv * hv);
    }
  }
  __syncthreads();
  if (threadIdx.x < 32) {
    atomicAdd(&sums[threadIdx.x], sSum[threadIdx.x]);
    atomicAdd(&sums[32 + threadIdx.x], sSq[threadIdx.x]);
  }
}

// ============================================================
// K5a/K5b: batch-norm stats and apply
// ============================================================
__global__ void bn_stats_kernel(const float* __restrict__ sums,
                                float* __restrict__ stats, int N) {
  int t = threadIdx.x;
  if (t < 32) {
    float mean = sums[t] / (float)N;
    float var  = sums[32 + t] / (float)N - mean * mean;
    stats[t]      = mean;
    stats[32 + t] = rsqrtf(var + 1e-5f);
  }
}

__global__ __launch_bounds__(256) void bn_apply_kernel(
    const float* __restrict__ hbuf, const float* __restrict__ stats,
    const float* __restrict__ gamma, const float* __restrict__ betab,
    float* __restrict__ out, int total) {
  int i = blockIdx.x * 256 + threadIdx.x;
  if (i >= total) return;
  int t = i & 31;
  out[i] = (hbuf[i] - stats[t]) * stats[32 + t] * gamma[t] + betab[t];
}

// ============================================================
extern "C" void kernel_launch(void* const* d_in, const int* in_sizes, int n_in,
                              void* d_out, int out_size, void* d_ws, size_t ws_size,
                              hipStream_t stream) {
  const float* x         = (const float*)d_in[0];
  const float* edge_attr = (const float*)d_in[1];
  const float* Wq = (const float*)d_in[2];  const float* bq = (const float*)d_in[3];
  const float* Wk = (const float*)d_in[4];  const float* bk = (const float*)d_in[5];
  const float* Wv = (const float*)d_in[6];  const float* bv = (const float*)d_in[7];
  const float* We = (const float*)d_in[8];
  const float* Wskip = (const float*)d_in[9]; const float* bskip = (const float*)d_in[10];
  const float* Wbeta = (const float*)d_in[11];
  const float* Wt = (const float*)d_in[12]; const float* bt = (const float*)d_in[13];
  const float* gamma = (const float*)d_in[14];
  const float* betab = (const float*)d_in[15];
  const int* eidx = (const int*)d_in[16];

  const int N = in_sizes[0] / 128;
  const int E = in_sizes[1];
  (void)n_in; (void)out_size; (void)ws_size;

  char* base = (char*)d_ws;
  size_t off = 0;
  auto carve = [&](size_t bytes) -> void* {
    void* p = base + off;
    off += (bytes + 255) & ~(size_t)255;
    return p;
  };
  float*    q      = (float*)carve((size_t)N * HCC * 4);
  float*    kk     = (float*)carve((size_t)N * HCC * 4);
  float*    vv     = (float*)carve((size_t)N * HCC * 4);
  float*    xr     = (float*)carve((size_t)N * HCC * 4);
  float*    outr   = (float*)carve((size_t)N * HCC * 4);
  float*    alpha  = (float*)carve((size_t)E * 3 * 4);
  unsigned* amax   = (unsigned*)carve((size_t)N * 3 * 4);
  float*    denom  = (float*)carve((size_t)N * 3 * 4);
  float*    hbuf   = (float*)carve((size_t)N * 32 * 4);
  float*    sums   = (float*)carve(64 * 4);
  float*    stats  = (float*)carve(64 * 4);
  __bf16*   packed = (__bf16*)carve((size_t)4 * 128 * 96 * 2);

  dim3 blk(256);
  pack_weights_kernel<<<(4 * 128 * 96 + 255) / 256, blk, 0, stream>>>(
      Wq, Wk, Wv, Wskip, packed);
  init_kernel<<<((size_t)N * HCC + 255) / 256, blk, 0, stream>>>(
      outr, amax, denom, sums, N);
  proj_wmma_kernel<<<(N + TILE_M - 1) / TILE_M, dim3(128), 0, stream>>>(
      x, packed, bq, bk, bv, bskip, q, kk, vv, xr, N);
  edge_alpha_kernel<<<(E + 255) / 256, blk, 0, stream>>>(
      q, kk, edge_attr, We, eidx, alpha, amax, E);
  edge_accum_kernel<<<(E + 255) / 256, blk, 0, stream>>>(
      vv, edge_attr, We, eidx, alpha, amax, denom, outr, E);
  node_finalize_kernel<<<(N + 255) / 256, blk, 0, stream>>>(
      outr, denom, xr, Wbeta, Wt, bt, hbuf, sums, N);
  bn_stats_kernel<<<1, 32, 0, stream>>>(sums, stats, N);
  bn_apply_kernel<<<((size_t)N * 32 + 255) / 256, blk, 0, stream>>>(
      hbuf, stats, gamma, betab, (float*)d_out, N * 32);
}
